// MultiheadAttention_32427003085519
// MI455X (gfx1250) — compile-verified, matrix operand order checked by reading, not run on hardware
//
#include <hip/hip_runtime.h>
#include <hip/hip_bf16.h>

typedef __attribute__((ext_vector_type(16))) _Float16 v16h;
typedef __attribute__((ext_vector_type(8)))  _Float16 v8h;
typedef __attribute__((ext_vector_type(8)))  float    v8f;
typedef __attribute__((ext_vector_type(4)))  float    v4f;

#define B_  4
#define S_  2048
#define D_  1024
#define H_  16
#define DH_ 64
#define M_  (B_ * S_)              // 8192 rows
#define MASK_VALUE (-1000000.0f)

static __device__ __forceinline__ v8f wmma_f16(v16h a, v16h b, v8f c) {
  // D = A(16x32 f16) * B(32x16 f16) + C(16x16 f32)
  return __builtin_amdgcn_wmma_f32_16x16x32_f16(false, a, false, b, (short)0, c,
                                                false, false);
}

// ---------------------------------------------------------------------------
// Fragment loaders. A-operand layout (16x32, f16):
//   lane = 16*half + m  (m = row 0..15)
//   elements 0..7  <- cols k0 + half*8      .. +7
//   elements 8..15 <- cols k0 + 16 + half*8 .. +7
// B-operand is supplied as B^T (N x K) with the identical layout.
// ---------------------------------------------------------------------------
static __device__ __forceinline__ v16h frag_ld_f16(const _Float16* __restrict__ src,
                                                   size_t row0, size_t ld, int k0) {
  const int lane = threadIdx.x & 31;
  const int m = lane & 15, half = lane >> 4;
  const _Float16* p = src + (row0 + (size_t)m) * ld + (size_t)(k0 + half * 8);
  v8h lo = *(const v8h*)(p);        // 16B aligned: ld, k0 multiples of 8
  v8h hi = *(const v8h*)(p + 16);
  v16h out;
#pragma unroll
  for (int j = 0; j < 8; ++j) { out[j] = lo[j]; out[j + 8] = hi[j]; }
  return out;
}

static __device__ __forceinline__ v16h frag_ld_f32(const float* __restrict__ src,
                                                   size_t row0, size_t ld, int k0) {
  const int lane = threadIdx.x & 31;
  const int m = lane & 15, half = lane >> 4;
  const float* p = src + (row0 + (size_t)m) * ld + (size_t)(k0 + half * 8);
  v4f a0 = *(const v4f*)(p);
  v4f a1 = *(const v4f*)(p + 4);
  v4f b0 = *(const v4f*)(p + 16);
  v4f b1 = *(const v4f*)(p + 20);
  v16h out;
#pragma unroll
  for (int j = 0; j < 4; ++j) {
    out[j]      = (_Float16)a0[j];
    out[j + 4]  = (_Float16)a1[j];
    out[j + 8]  = (_Float16)b0[j];
    out[j + 12] = (_Float16)b1[j];
  }
  return out;
}

// Build a fragment from two 16-byte LDS vectors (compiler emits ds_load_b128).
static __device__ __forceinline__ v16h frag_pack(const v8h lo, const v8h hi) {
  v16h out;
#pragma unroll
  for (int j = 0; j < 8; ++j) { out[j] = lo[j]; out[j + 8] = hi[j]; }
  return out;
}

// ---------------------------------------------------------------------------
// GEMM: C[m,n] = sum_k A[m,k] * W[n,k]   (M=8192, N=K=1024)
// OUTMODE 0: f16 out in [B,H,S,DH];  1: f16 out in [B,H,DH,S] (V^T);
//         2: f32 out row-major [M,N] (final result).
// Block = 256 threads = 8 waves; wave tile 32x32 (2x2 WMMA); block tile 128x64.
// ---------------------------------------------------------------------------
template <bool A_F16, int OUTMODE>
__global__ __launch_bounds__(256) void gemm_kernel(const void* __restrict__ Aptr,
                                                   const float* __restrict__ W,
                                                   void* __restrict__ Out) {
  const int wave = threadIdx.x >> 5;
  const int lane = threadIdx.x & 31;
  const int mBlk = (blockIdx.x % (M_ / 128)) * 128;
  const int nBlk = (blockIdx.x / (M_ / 128)) * 64;
  const size_t m0 = mBlk + (wave & 3) * 32;
  const size_t n0 = nBlk + (wave >> 2) * 32;

  v8f acc[2][2];
#pragma unroll
  for (int i = 0; i < 2; ++i)
#pragma unroll
    for (int j = 0; j < 2; ++j)
#pragma unroll
      for (int r = 0; r < 8; ++r) acc[i][j][r] = 0.0f;

  for (int k0 = 0; k0 < D_; k0 += 32) {
    v16h a0, a1;
    if (A_F16) {
      a0 = frag_ld_f16((const _Float16*)Aptr, m0, D_, k0);
      a1 = frag_ld_f16((const _Float16*)Aptr, m0 + 16, D_, k0);
    } else {
      a0 = frag_ld_f32((const float*)Aptr, m0, D_, k0);
      a1 = frag_ld_f32((const float*)Aptr, m0 + 16, D_, k0);
    }
    v16h b0 = frag_ld_f32(W, n0, D_, k0);
    v16h b1 = frag_ld_f32(W, n0 + 16, D_, k0);
    acc[0][0] = wmma_f16(a0, b0, acc[0][0]);
    acc[0][1] = wmma_f16(a0, b1, acc[0][1]);
    acc[1][0] = wmma_f16(a1, b0, acc[1][0]);
    acc[1][1] = wmma_f16(a1, b1, acc[1][1]);
  }

  const int half = lane >> 4, ln = lane & 15;
#pragma unroll
  for (int tm = 0; tm < 2; ++tm)
#pragma unroll
    for (int tn = 0; tn < 2; ++tn)
#pragma unroll
      for (int r = 0; r < 8; ++r) {
        const int m = (int)m0 + tm * 16 + r + 8 * half;
        const int n = (int)n0 + tn * 16 + ln;
        const float v = acc[tm][tn][r];
        if (OUTMODE == 2) {
          ((float*)Out)[(size_t)m * D_ + n] = v;
        } else {
          const int b = m / S_, s = m % S_, h = n / DH_, d = n % DH_;
          const size_t bh = (size_t)b * H_ + h;
          const size_t idx = (OUTMODE == 0) ? (bh * S_ + s) * DH_ + d
                                            : (bh * DH_ + d) * S_ + s;
          ((_Float16*)Out)[idx] = (_Float16)v;
        }
      }
}

// ---------------------------------------------------------------------------
// Flash attention: one block per (b, h, 128-query tile); each wave owns 16
// query rows. K/V chunks (32 keys) are staged block-wide into double-buffered
// LDS with gfx1250 async global->LDS copies (ASYNCcnt), overlapping the next
// chunk's copy with the current chunk's WMMAs + online softmax.
// Q,K in [B,H,S,DH] f16; V in [B,H,DH,S] f16 (transposed); out O in [B,S,D] f16.
// ---------------------------------------------------------------------------
__global__ __launch_bounds__(256) void flash_attn_kernel(
    const _Float16* __restrict__ Qh, const _Float16* __restrict__ Kh,
    const _Float16* __restrict__ Vt, const int* __restrict__ valid_lens,
    _Float16* __restrict__ O) {
  __shared__ _Float16 Kbuf[2][32][DH_];   // 2 x 4KB : 32 keys x 64 dh
  __shared__ _Float16 Vbuf[2][DH_][32];   // 2 x 4KB : 64 dh  x 32 keys (V^T rows)
  __shared__ _Float16 Pbuf[8][16][32];    // 8KB per-wave probability staging

  const int tid = threadIdx.x;
  const int wave = tid >> 5;
  const int lane = tid & 31;
  const int half = lane >> 4;
  const int ln = lane & 15;
  const int qt = blockIdx.x & (S_ / 128 - 1);   // 16 query tiles
  const int bh = blockIdx.x >> 4;
  const int b = bh >> 4;                        // H_ == 16
  const int h = bh & (H_ - 1);
  const int vl = valid_lens[b];
  const float scale = 0.125f;                   // 1/sqrt(DH)

  const size_t kBase = (size_t)bh * S_;
  const size_t vBase = (size_t)bh * DH_;

  // Per-thread slices for the cooperative async copies (one B128 each).
  const int krow = tid >> 3, kcol = (tid & 7) * 8;   // K: 32 rows x 64 halves
  const int vrow = tid >> 2, vcol = (tid & 3) * 8;   // V: 64 rows x 32 halves

  auto stage = [&](int kc, int bi) {
    const _Float16* gk = Kh + (kBase + kc + krow) * DH_ + kcol;
    unsigned lk = (unsigned)(size_t)&Kbuf[bi][krow][kcol];
    asm volatile("global_load_async_to_lds_b128 %0, %1, off"
                 :: "v"(lk), "v"(gk) : "memory");
    const _Float16* gv = Vt + (vBase + vrow) * S_ + kc + vcol;
    unsigned lv = (unsigned)(size_t)&Vbuf[bi][vrow][vcol];
    asm volatile("global_load_async_to_lds_b128 %0, %1, off"
                 :: "v"(lv), "v"(gv) : "memory");
  };

  const size_t qRow = (size_t)bh * S_ + qt * 128 + wave * 16;
  const v16h qa0 = frag_ld_f16(Qh, qRow, DH_, 0);
  const v16h qa1 = frag_ld_f16(Qh, qRow, DH_, 32);

  float rmax[8], rsum[8];
  v8f oc[4];
#pragma unroll
  for (int r = 0; r < 8; ++r) { rmax[r] = -3.0e38f; rsum[r] = 0.0f; }
#pragma unroll
  for (int t = 0; t < 4; ++t)
#pragma unroll
    for (int r = 0; r < 8; ++r) oc[t][r] = 0.0f;

  stage(0, 0);  // prime first buffer

  for (int kc = 0; kc < S_; kc += 32) {
    const int bi = (kc >> 5) & 1;
    asm volatile("s_wait_asynccnt 0x0" ::: "memory");  // own async copies done
    __syncthreads();                                   // everyone's copies done
    if (kc + 32 < S_) stage(kc + 32, bi ^ 1);          // overlap next chunk

    // ---- scores: two 16x16 tiles over DH=64 (K-split 32+32), K from LDS ----
    v8f s0, s1;
#pragma unroll
    for (int r = 0; r < 8; ++r) { s0[r] = 0.0f; s1[r] = 0.0f; }
    {
      v16h kb;
      kb = frag_pack(*(const v8h*)&Kbuf[bi][ln][half * 8],
                     *(const v8h*)&Kbuf[bi][ln][half * 8 + 16]);
      s0 = wmma_f16(qa0, kb, s0);
      kb = frag_pack(*(const v8h*)&Kbuf[bi][ln][32 + half * 8],
                     *(const v8h*)&Kbuf[bi][ln][32 + half * 8 + 16]);
      s0 = wmma_f16(qa1, kb, s0);
      kb = frag_pack(*(const v8h*)&Kbuf[bi][16 + ln][half * 8],
                     *(const v8h*)&Kbuf[bi][16 + ln][half * 8 + 16]);
      s1 = wmma_f16(qa0, kb, s1);
      kb = frag_pack(*(const v8h*)&Kbuf[bi][16 + ln][32 + half * 8],
                     *(const v8h*)&Kbuf[bi][16 + ln][32 + half * 8 + 16]);
      s1 = wmma_f16(qa1, kb, s1);
    }

    // ---- scale + key mask + online softmax (row = vgpr r + 8*half) ----
    const bool ok0 = (kc + ln) < vl;
    const bool ok1 = (kc + 16 + ln) < vl;
    float p0[8], p1[8], sf[8];
#pragma unroll
    for (int r = 0; r < 8; ++r) {
      const float f0 = ok0 ? s0[r] * scale : MASK_VALUE;
      const float f1 = ok1 ? s1[r] * scale : MASK_VALUE;
      float cm = fmaxf(f0, f1);
      cm = fmaxf(cm, __shfl_xor(cm, 1));
      cm = fmaxf(cm, __shfl_xor(cm, 2));
      cm = fmaxf(cm, __shfl_xor(cm, 4));
      cm = fmaxf(cm, __shfl_xor(cm, 8));
      const float nm = fmaxf(rmax[r], cm);
      const float fs = __expf(rmax[r] - nm);
      rmax[r] = nm;
      const float e0 = __expf(f0 - nm);
      const float e1 = __expf(f1 - nm);
      float rs = e0 + e1;
      rs += __shfl_xor(rs, 1);
      rs += __shfl_xor(rs, 2);
      rs += __shfl_xor(rs, 4);
      rs += __shfl_xor(rs, 8);
      rsum[r] = rsum[r] * fs + rs;
      p0[r] = e0; p1[r] = e1; sf[r] = fs;
    }
#pragma unroll
    for (int t = 0; t < 4; ++t)
#pragma unroll
      for (int r = 0; r < 8; ++r) oc[t][r] *= sf[r];

    // ---- reshape P: C-layout -> A-fragment via per-wave LDS ----
#pragma unroll
    for (int r = 0; r < 8; ++r) {
      Pbuf[wave][r + 8 * half][ln]      = (_Float16)p0[r];
      Pbuf[wave][r + 8 * half][16 + ln] = (_Float16)p1[r];
    }
    const v16h pf = frag_pack(*(const v8h*)&Pbuf[wave][ln][half * 8],
                              *(const v8h*)&Pbuf[wave][ln][half * 8 + 16]);

    // ---- PV: out(16x64) += P(16x32) * V(32x64), V^T rows from LDS ----
#pragma unroll
    for (int t = 0; t < 4; ++t) {
      const v16h vf = frag_pack(*(const v8h*)&Vbuf[bi][t * 16 + ln][half * 8],
                                *(const v8h*)&Vbuf[bi][t * 16 + ln][half * 8 + 16]);
      oc[t] = wmma_f16(pf, vf, oc[t]);
    }
  }

  // ---- normalize and store O[b, q, h*DH + d] ----
  const int q0 = qt * 128 + wave * 16;
#pragma unroll
  for (int t = 0; t < 4; ++t)
#pragma unroll
    for (int r = 0; r < 8; ++r) {
      const int q = q0 + r + 8 * half;
      const size_t idx = ((size_t)b * S_ + q) * D_ + h * DH_ + t * 16 + ln;
      O[idx] = (_Float16)(oc[t][r] / rsum[r]);
    }
}

// ---------------------------------------------------------------------------
extern "C" void kernel_launch(void* const* d_in, const int* in_sizes, int n_in,
                              void* d_out, int out_size, void* d_ws, size_t ws_size,
                              hipStream_t stream) {
  const float* key   = (const float*)d_in[0];
  const float* query = (const float*)d_in[1];
  const float* value = (const float*)d_in[2];
  const int*   vlens = (const int*)d_in[3];
  const float* Wk    = (const float*)d_in[4];
  const float* Wq    = (const float*)d_in[5];
  const float* Wv    = (const float*)d_in[6];
  const float* Wo    = (const float*)d_in[7];
  float* out = (float*)d_out;

  char* ws = (char*)d_ws;
  const size_t bufBytes = (size_t)B_ * H_ * S_ * DH_ * sizeof(_Float16);  // 16 MB
  _Float16* Qh = (_Float16*)(ws);
  _Float16* Kh = (_Float16*)(ws + bufBytes);
  _Float16* Vt = (_Float16*)(ws + 2 * bufBytes);
  _Float16* O  = (_Float16*)(ws + 3 * bufBytes);

  const dim3 blk(256);
  const dim3 gGemm((M_ / 128) * (D_ / 64));          // 1024 blocks
  const dim3 gAttn(B_ * H_ * (S_ / 128));            // 1024 blocks

  gemm_kernel<false, 0><<<gGemm, blk, 0, stream>>>((const void*)query, Wq, (void*)Qh);
  gemm_kernel<false, 0><<<gGemm, blk, 0, stream>>>((const void*)key,   Wk, (void*)Kh);
  gemm_kernel<false, 1><<<gGemm, blk, 0, stream>>>((const void*)value, Wv, (void*)Vt);
  flash_attn_kernel<<<gAttn, blk, 0, stream>>>(Qh, Kh, Vt, vlens, O);
  gemm_kernel<true, 2><<<gGemm, blk, 0, stream>>>((const void*)O, Wo, (void*)out);
}
